// MinkOccupancyForecastingNetwork3D_11493332484203
// MI455X (gfx1250) — compile-verified, matrix operand order checked by reading, not run on hardware
//
#include <hip/hip_runtime.h>
#include <hip/hip_bf16.h>
#include <math.h>

// ---------------------------------------------------------------------------
// MinkOccupancyForecastingNetwork3D forward on gfx1250 (CDNA5, wave32, WMMA)
// Grid: X=192, Y=192, Z=32, T=2.  Layout NCDHW (B=1): flat = ((c*192+x)*192+y)*32+z
// Conv = implicit GEMM on v_wmma_f32_16x16x32_f16, templated on <KS,STRIDE>,
// K re-ordered (kernel-offset major, channel minor, channel padded to
// CP=2^k >= max(Cin,8)), weights pre-packed into A-fragment layout.
// All flat-index decompositions use power-of-two OW (true for every layer
// here) + blockIdx.y/z for od / channel -> no integer division anywhere hot.
// ---------------------------------------------------------------------------

typedef __attribute__((ext_vector_type(16))) _Float16 v16h;
typedef __attribute__((ext_vector_type(8)))  float    v8f;

#define GX 192
#define GY 192
#define GZ 32
#define NS 128

// ------------------------------- utility kernels ---------------------------

__global__ void k_zero(float* __restrict__ p, long long n) {
  long long i = (long long)blockIdx.x * blockDim.x + threadIdx.x;
  if (i < n) p[i] = 0.0f;
}

__global__ void k_voxelize(const float* __restrict__ pts, float* __restrict__ feat,
                           float* __restrict__ m1, int N) {
  int i = blockIdx.x * blockDim.x + threadIdx.x;
  if (i >= N) return;
  float px = pts[4 * i + 0], py = pts[4 * i + 1], pz = pts[4 * i + 2], pt = pts[4 * i + 3];
  int ix = min(max((int)floorf((px + 19.2f) / 0.2f), 0), GX - 1);
  int iy = min(max((int)floorf((py + 19.2f) / 0.2f), 0), GY - 1);
  int iz = min(max((int)floorf((pz + 3.2f) / 0.2f), 0), GZ - 1);
  int ti = min(max((int)pt, 0), 1);
  size_t sp = ((size_t)ix * GY + iy) * GZ + iz;
  feat[(size_t)ti * GX * GY * GZ + sp] = 1.0f;   // one-hot max with 1.0: plain store is fine
  m1[sp] = 1.0f;
}

// mask 2x downsample; blockIdx.y = od; plane decode by shift/mask (OW/2 pow2)
__global__ void k_pool2(const float* __restrict__ m, float* __restrict__ o,
                        int H, int W, int lg2ow) {
  int OH = H / 2, OW = W / 2;
  int p = blockIdx.x * blockDim.x + threadIdx.x;
  if (p >= OH * OW) return;
  int od = blockIdx.y;
  int oh = p >> lg2ow, ow = p & (OW - 1);
  float v = 0.0f;
  for (int dz = 0; dz < 2; ++dz)
    for (int dy = 0; dy < 2; ++dy)
      for (int dx = 0; dx < 2; ++dx)
        v = fmaxf(v, m[(((size_t)(od * 2 + dz)) * H + (oh * 2 + dy)) * W + (ow * 2 + dx)]);
  o[(size_t)od * OH * OW + p] = v;
}

// masked-BN stats: one block per channel; binary mask => sum((x m)^2) == sum(x^2 m)
__global__ void k_bn_stats(const float* __restrict__ x, const float* __restrict__ m,
                           float* __restrict__ st, int n) {
  int c = blockIdx.x;
  const float* xc = x + (size_t)c * n;
  float s = 0.f, q = 0.f, cnt = 0.f;
  for (int i = threadIdx.x; i < n; i += blockDim.x) {
    float mv = m[i];
    float v = xc[i] * mv;
    s += v; q += v * v; cnt += mv;
  }
  __shared__ float sh0[256], sh1[256], sh2[256];
  int tid = threadIdx.x;
  sh0[tid] = s; sh1[tid] = q; sh2[tid] = cnt;
  __syncthreads();
  for (int o = 128; o > 0; o >>= 1) {
    if (tid < o) { sh0[tid] += sh0[tid + o]; sh1[tid] += sh1[tid + o]; sh2[tid] += sh2[tid + o]; }
    __syncthreads();
  }
  if (tid == 0) { st[3 * c] = sh0[0]; st[3 * c + 1] = sh1[0]; st[3 * c + 2] = sh2[0]; }
}

// blockIdx.y = channel -> no division
__global__ void k_bn_apply(const float* __restrict__ x, const float* __restrict__ m,
                           const float* __restrict__ st, const float* __restrict__ g,
                           const float* __restrict__ b, float* __restrict__ y,
                           int n, int relu) {
  int i = blockIdx.x * blockDim.x + threadIdx.x;
  if (i >= n) return;
  int c = blockIdx.y;
  size_t e = (size_t)c * n + i;
  float cnt = fmaxf(st[3 * c + 2], 1.0f);
  float mu  = st[3 * c] / cnt;
  float var = st[3 * c + 1] / cnt - mu * mu;
  float v = (x[e] - mu) * rsqrtf(var + 1e-5f);
  v = (v * g[c] + b[c]) * m[i];
  if (relu) v = fmaxf(v, 0.0f);
  y[e] = v;
}

__global__ void k_add_relu(const float* __restrict__ a, const float* __restrict__ b,
                           float* __restrict__ y, long long n) {
  long long i = (long long)blockIdx.x * blockDim.x + threadIdx.x;
  if (i < n) y[i] = fmaxf(a[i] + b[i], 0.0f);
}

__global__ void k_bias_mask(float* __restrict__ x, const float* __restrict__ bias,
                            const float* __restrict__ m, int n) {
  int i = blockIdx.x * blockDim.x + threadIdx.x;
  if (i >= n) return;
  int c = blockIdx.y;
  size_t e = (size_t)c * n + i;
  x[e] = (x[e] + bias[c]) * m[i];
}

// 2x2x2 stride-2 transpose conv (exact upsample). Weight (ci,co,2,2,2).
// blockIdx.y = od, blockIdx.z = co; plane decode by shift/mask (OW=2W pow2).
__global__ void k_convtr(const float* __restrict__ in, const float* __restrict__ w,
                         float* __restrict__ out, int Cin, int Cout,
                         int D, int H, int W, int lg2ow) {
  int OH = 2 * H, OW = 2 * W;
  int p = blockIdx.x * blockDim.x + threadIdx.x;
  if (p >= OH * OW) return;
  int od = blockIdx.y;
  int co = blockIdx.z;
  int oh = p >> lg2ow, ow = p & (OW - 1);
  int id = od >> 1, ih = oh >> 1, iw = ow >> 1;
  int off = ((od & 1) << 2) | ((oh & 1) << 1) | (ow & 1);
  float acc = 0.0f;
  for (int ci = 0; ci < Cin; ++ci)
    acc += in[(((size_t)ci * D + id) * H + ih) * W + iw] * w[((size_t)ci * Cout + co) * 8 + off];
  out[(size_t)co * (8 * (size_t)D * H * W) + (size_t)od * OH * OW + p] = acc;
}

// --------------------- A-fragment pre-pack (f32 -> f16) ---------------------
// K' ordering: k' = offIdx*CP + ci  (offIdx = kd*ks^2+kh*ks+kw, CP=1<<lg2cp)
// Fragment striping (16-bit matrices): lane's half h holds chunk position
//   cpos = h + hiHalf + (h>=8 ? 8 : 0),  hiHalf = (lane&16)?8:0
// Packed layout: apk[((ct*nChunks + chunk)*32 + lane)*16 + h]
// grid = (nChunks, ctTiles), block = 32  -> no div/mod.
__global__ __launch_bounds__(32) void k_pack_a(const float* __restrict__ w,
                                               _Float16* __restrict__ apk,
                                               int Cin, int Cout, int ks, int lg2cp,
                                               int nChunks) {
  int lane  = threadIdx.x;
  int chunk = blockIdx.x;
  int ct    = blockIdx.y;
  int CP  = 1 << lg2cp;
  int ks3 = ks * ks * ks;
  int K   = Cin * ks3;
  int hiHalf = (lane & 16) ? 8 : 0;
  int m  = lane & 15;
  int co = ct * 16 + m;
  v16h vals;
#pragma unroll
  for (int h = 0; h < 16; ++h) {
    int cpos = h + hiHalf + ((h >= 8) ? 8 : 0);
    int kp = chunk * 32 + cpos;
    int offIdx = kp >> lg2cp;
    int ci = kp & (CP - 1);
    float v = 0.0f;
    if (co < Cout && ci < Cin && offIdx < ks3)
      v = w[(size_t)co * K + (size_t)ci * ks3 + offIdx];
    vals[h] = (_Float16)v;
  }
  *(v16h*)(apk + ((size_t)((size_t)ct * nChunks + chunk) * 32 + lane) * 16) = vals;
}

// ------------------------- WMMA implicit-GEMM conv -------------------------
// One wave per 16(co) x 16(vox) tile.  grid = (planeTiles, OD, ctTiles).
// A = packed weights (single v16h load / chunk).  B = im2col: per chunk each
// lane fills two runs of 8 consecutive k' (one kernel offset each, constant
// stride DHW).  KS/STRIDE compile-time -> offset decompose is shifts/mads.
template <int KS, int STRIDE>
__global__ __launch_bounds__(32) void k_conv_wmma(
    const float* __restrict__ in, const _Float16* __restrict__ apk,
    float* __restrict__ out, int Cin, int Cout, int D, int H, int W,
    int lg2cp, int nChunks, int OH, int OW, int lg2ow) {
  constexpr int PAD = (STRIDE == 1) ? (KS - 1) / 2 : 0;
  constexpr int KS2 = KS * KS, KS3 = KS * KS * KS;

  int plane = OH * OW;
  int lane = threadIdx.x;
  int hiHalf = (lane & 16) ? 8 : 0;

  int n = lane & 15;
  int p = blockIdx.x * 16 + n;            // voxel within (oh,ow) plane
  bool voxOK = p < plane;
  int pp = voxOK ? p : 0;
  int oh = pp >> lg2ow, ow = pp & (OW - 1);
  int od = blockIdx.y;
  int ct = blockIdx.z;

  int CP = 1 << lg2cp;
  size_t DHW = (size_t)D * H * W;

  const _Float16* ap = apk + ((size_t)((size_t)ct * nChunks) * 32 + lane) * 16;

  v8f acc = {};
#pragma unroll 2
  for (int c = 0; c < nChunks; ++c) {
    v16h a = *(const v16h*)ap;
    __builtin_prefetch((const void*)(ap + 1024), 0, 3);   // global_prefetch_b8: A two chunks ahead
    ap += 512;
    int kb = c << 5;
    v16h b;
#pragma unroll
    for (int g = 0; g < 2; ++g) {
      int s = kb + hiHalf + g * 16;        // start of 8 consecutive k'
      int offIdx = s >> lg2cp;
      int ciB = s & (CP - 1);
      int r = offIdx;
      int kd = r / KS2; r -= kd * KS2;     // constant divisors -> shifts/mads
      int kh = r / KS;  int kw = r - kh * KS;
      int id = od * STRIDE - PAD + kd;
      int ih = oh * STRIDE - PAD + kh;
      int iw = ow * STRIDE - PAD + kw;
      bool valid = voxOK && (offIdx < KS3) &&
                   id >= 0 && id < D && ih >= 0 && ih < H && iw >= 0 && iw < W;
      const float* bp = in + (size_t)ciB * DHW + (((size_t)id * H + ih) * W + iw);
#pragma unroll
      for (int j = 0; j < 8; ++j) {
        float bv = (valid && (ciB + j) < Cin) ? bp[(size_t)j * DHW] : 0.0f;
        b[g * 8 + j] = (_Float16)bv;
      }
    }
    acc = __builtin_amdgcn_wmma_f32_16x16x32_f16(false, a, false, b, (short)0, acc,
                                                 false, false);
  }

  if (voxOK) {
    size_t nV = (size_t)gridDim.y * plane;
    size_t sp = (size_t)od * plane + p;
    int co0 = ct * 16;
#pragma unroll
    for (int v = 0; v < 8; ++v) {
      int m = v + ((lane & 16) ? 8 : 0);
      int co = co0 + m;
      if (co < Cout) out[(size_t)co * nV + sp] = acc[v];
    }
  }
}

// ------------------------------- render ------------------------------------

__global__ void k_render(const float* __restrict__ fin, const float* __restrict__ origin,
                         const float* __restrict__ pts, const int* __restrict__ tidx,
                         float* __restrict__ pred, int N) {
  int i = blockIdx.x * blockDim.x + threadIdx.x;
  if (i >= N) return;
  int t = min(max(tidx[i], 0), 1);
  float ox = (origin[t * 3 + 0] + 19.2f) / 0.2f;
  float oy = (origin[t * 3 + 1] + 19.2f) / 0.2f;
  float oz = (origin[t * 3 + 2] + 3.2f) / 0.2f;
  float px = (pts[3 * i + 0] + 19.2f) / 0.2f;
  float py = (pts[3 * i + 1] + 19.2f) / 0.2f;
  float pz = (pts[3 * i + 2] + 3.2f) / 0.2f;
  float dx = px - ox, dy = py - oy, dz = pz - oz;
  float gt = sqrtf(dx * dx + dy * dy + dz * dz);
  float inv = 1.0f / fmaxf(gt, 1e-6f);
  dx *= inv; dy *= inv; dz *= inv;
  float dt = gt / (float)NS;
  const float* ft = fin + (size_t)t * GX * GY * GZ;
  float tau = 0.0f, acc = 0.0f;
  for (int s = 0; s < NS; ++s) {
    float tv = gt * ((s + 0.5f) / (float)NS);
    float qx = ox + dx * tv, qy = oy + dy * tv, qz = oz + dz * tv;
    int ix = (int)floorf(qx), iy = (int)floorf(qy), iz = (int)floorf(qz);
    float sig = 0.0f;
    if (ix >= 0 && ix < GX && iy >= 0 && iy < GY && iz >= 0 && iz < GZ)
      sig = fmaxf(ft[((size_t)ix * GY + iy) * GZ + iz], 0.0f);  // relu(sigma)
    float a = sig * dt;
    acc += expf(-tau) * (1.0f - expf(-a)) * tv;                 // exclusive transmittance
    tau += a;
  }
  pred[i] = acc + expf(-tau) * gt;
}

// ------------------------------- host side ---------------------------------

struct Blk {
  const float *b1, *b2, *bd, *g1, *g2, *gd, *w1, *w2, *wd;
};

static inline int gs(long long n, int bs) { return (int)((n + bs - 1) / bs); }
static inline int lg2i(int x) { int l = 0; while ((1 << l) < x) ++l; return l; }

extern "C" void kernel_launch(void* const* d_in, const int* in_sizes, int n_in,
                              void* d_out, int out_size, void* d_ws, size_t ws_size,
                              hipStream_t stream) {
  (void)n_in; (void)ws_size;
  const int n1 = GX * GY * GZ;        // 1179648
  const int n2 = 96 * 96 * 16;        // 147456
  const int n4 = 48 * 48 * 8;         // 18432
  const int n8 = 24 * 24 * 4;         // 2304
  const int n16 = 12 * 12 * 2;        // 288

  // ---- inputs ----
  const float* in_pts  = (const float*)d_in[0];   // [1,100000,4]
  const float* origin  = (const float*)d_in[1];   // [1,2,3]
  const float* out_pts = (const float*)d_in[2];   // [1,50000,3]
  const int*   tindex  = (const int*)d_in[3];     // [1,50000]
  int Npts  = in_sizes[0] / 4;
  int Nrays = in_sizes[3];

  // ---- params (jax pytree flatten order: sorted dict keys, recursively) ----
  int idx = 4;
  auto f = [&](void) { return (const float*)d_in[idx++]; };
  const float *b0 = f(), *b1 = f(), *b2 = f(), *b3 = f(), *b4 = f(), *bf = f();
  auto readBlk = [&](bool hasWd) {
    Blk P{};
    P.b1 = f(); P.b2 = f(); if (hasWd) P.bd = f();
    P.g1 = f(); P.g2 = f(); if (hasWd) P.gd = f();
    P.w1 = f(); P.w2 = f(); if (hasWd) P.wd = f();
    return P;
  };
  Blk K1 = readBlk(false), K2 = readBlk(true), K3 = readBlk(true), K4 = readBlk(true);
  Blk K5 = readBlk(true),  K6 = readBlk(true), K7 = readBlk(true), K8 = readBlk(true);
  const float *bt4 = f(), *bt5 = f(), *bt6 = f(), *bt7 = f();
  const float *g0 = f(), *g1 = f(), *g2 = f(), *g3 = f(), *g4 = f();
  const float *gt4 = f(), *gt5 = f(), *gt6 = f(), *gt7 = f();
  const float *w0 = f(), *w1 = f(), *w2 = f(), *w3 = f(), *w4 = f();
  const float *wf = f();
  const float *wt4 = f(), *wt5 = f(), *wt6 = f(), *wt7 = f();

  // ---- workspace bump allocator (all sizes multiples of 8 floats -> 32B align) ----
  float* WS = (float*)d_ws;
  size_t off = 0;
  auto alloc = [&](size_t nf) { float* p = WS + off; off += nf; return p; };
  float* feat = alloc((size_t)2 * n1);
  float* m1   = alloc(n1);
  float* m2   = alloc(n2);
  float* m4   = alloc(n4);
  float* m8   = alloc(n8);
  float* m16  = alloc(n16);
  float* stats = alloc(3 * 128);
  _Float16* apk = (_Float16*)alloc(160 * 1024);   // 320K halves; max layer needs 221184
  float* out_p1 = alloc((size_t)8 * n1);
  float* out_b1 = alloc((size_t)8 * n2);
  float* out_b2 = alloc((size_t)16 * n4);
  float* out_b3 = alloc((size_t)32 * n8);
  float* fin = alloc((size_t)2 * n1);
  float* X1 = alloc((size_t)16 * n1); float* O1 = alloc((size_t)8 * n1);  float* T1 = alloc((size_t)16 * n1);
  float* X2 = alloc((size_t)24 * n2); float* O2 = alloc((size_t)16 * n2); float* T2 = alloc((size_t)32 * n2);
  float* X4 = alloc((size_t)48 * n4); float* O4 = alloc((size_t)32 * n4); float* T4 = alloc((size_t)64 * n4);
  float* X8 = alloc((size_t)96 * n8); float* O8 = alloc((size_t)64 * n8); float* T8 = alloc((size_t)128 * n8);
  float* X16 = alloc((size_t)32 * n16); float* O16 = alloc((size_t)64 * n16); float* T16 = alloc((size_t)128 * n16);

  // ---- helper launchers ----
  auto conv = [&](const float* in, const float* wgt, float* out,
                  int Cin, int Cout, int D, int H, int Wd, int ks, int stride) {
    int CP = 8, lg = 3;
    while (CP < Cin) { CP <<= 1; ++lg; }
    int ks3 = ks * ks * ks;
    int Kp = (ks3 * CP + 31) & ~31;
    int nChunks = Kp >> 5;
    int ctTiles = (Cout + 15) / 16;
    k_pack_a<<<dim3(nChunks, ctTiles), 32, 0, stream>>>(wgt, apk, Cin, Cout, ks, lg, nChunks);
    int OD = (stride == 1) ? D  : (D  - ks) / stride + 1;
    int OH = (stride == 1) ? H  : (H  - ks) / stride + 1;
    int OW = (stride == 1) ? Wd : (Wd - ks) / stride + 1;   // always pow2 in this net
    int planeTiles = (OH * OW + 15) / 16;
    dim3 grid(planeTiles, OD, ctTiles);
    int lw = lg2i(OW);
    if (ks == 3)
      k_conv_wmma<3, 1><<<grid, 32, 0, stream>>>(in, apk, out, Cin, Cout, D, H, Wd, lg, nChunks, OH, OW, lw);
    else if (ks == 1)
      k_conv_wmma<1, 1><<<grid, 32, 0, stream>>>(in, apk, out, Cin, Cout, D, H, Wd, lg, nChunks, OH, OW, lw);
    else if (ks == 2)
      k_conv_wmma<2, 2><<<grid, 32, 0, stream>>>(in, apk, out, Cin, Cout, D, H, Wd, lg, nChunks, OH, OW, lw);
    else
      k_conv_wmma<5, 1><<<grid, 32, 0, stream>>>(in, apk, out, Cin, Cout, D, H, Wd, lg, nChunks, OH, OW, lw);
  };
  auto bn = [&](float* x, const float* m, const float* g, const float* b,
                int C, int n, bool relu) {
    k_bn_stats<<<C, 256, 0, stream>>>(x, m, stats, n);
    k_bn_apply<<<dim3(gs(n, 256), C), 256, 0, stream>>>(x, m, stats, g, b, x, n, relu ? 1 : 0);
  };
  auto blockf = [&](const float* x, const Blk& P, const float* m, int Cin, int Cout,
                    int D, int H, int Wd, float* out, float* t1, float* t2) {
    int n = D * H * Wd;
    conv(x, P.w1, t1, Cin, Cout, D, H, Wd, 3, 1);  bn(t1, m, P.g1, P.b1, Cout, n, true);
    conv(t1, P.w2, t2, Cout, Cout, D, H, Wd, 3, 1); bn(t2, m, P.g2, P.b2, Cout, n, false);
    const float* res = x;
    if (P.wd) {
      conv(x, P.wd, out, Cin, Cout, D, H, Wd, 1, 1);
      bn(out, m, P.gd, P.bd, Cout, n, false);
      res = out;
    }
    k_add_relu<<<gs((long long)Cout * n, 256), 256, 0, stream>>>(t2, res, out, (long long)Cout * n);
  };
  auto convtr = [&](const float* in, const float* wgt, float* out,
                    int Cin, int Cout, int D, int H, int Wd) {
    dim3 grid(gs(4 * H * Wd, 256), 2 * D, Cout);
    k_convtr<<<grid, 256, 0, stream>>>(in, wgt, out, Cin, Cout, D, H, Wd, lg2i(2 * Wd));
  };
  auto pool = [&](const float* mi, float* mo, int D, int H, int Wd) {
    dim3 grid(gs((H / 2) * (Wd / 2), 256), D / 2);
    k_pool2<<<grid, 256, 0, stream>>>(mi, mo, H, Wd, lg2i(Wd / 2));
  };

  // ---- voxelize + mask pyramid ----
  k_zero<<<gs((long long)3 * n1, 256), 256, 0, stream>>>(feat, (long long)3 * n1); // feat+m1 adjacent
  k_voxelize<<<gs(Npts, 256), 256, 0, stream>>>(in_pts, feat, m1, Npts);
  pool(m1, m2, GX, GY, GZ);
  pool(m2, m4, 96, 96, 16);
  pool(m4, m8, 48, 48, 8);
  pool(m8, m16, 24, 24, 4);

  // ---- encoder ----
  conv(feat, w0, out_p1, 2, 8, GX, GY, GZ, 5, 1);      bn(out_p1, m1, g0, b0, 8, n1, true);
  conv(out_p1, w1, X2, 8, 8, GX, GY, GZ, 2, 2);        bn(X2, m2, g1, b1, 8, n2, true);
  blockf(X2, K1, m2, 8, 8, 96, 96, 16, out_b1, T2, T2 + (size_t)8 * n2);
  conv(out_b1, w2, X4, 8, 8, 96, 96, 16, 2, 2);        bn(X4, m4, g2, b2, 8, n4, true);
  blockf(X4, K2, m4, 8, 16, 48, 48, 8, out_b2, T4, T4 + (size_t)16 * n4);
  conv(out_b2, w3, X8, 16, 16, 48, 48, 8, 2, 2);       bn(X8, m8, g3, b3, 16, n8, true);
  blockf(X8, K3, m8, 16, 32, 24, 24, 4, out_b3, T8, T8 + (size_t)32 * n8);
  conv(out_b3, w4, X16, 32, 32, 24, 24, 4, 2, 2);      bn(X16, m16, g4, b4, 32, n16, true);
  blockf(X16, K4, m16, 32, 64, 12, 12, 2, O16, T16, T16 + (size_t)64 * n16);

  // ---- decoder (upsample, bn-relu, concat skip, block) ----
  convtr(O16, wt4, X8, 64, 64, 12, 12, 2);             bn(X8, m8, gt4, bt4, 64, n8, true);
  hipMemcpyAsync(X8 + (size_t)64 * n8, out_b3, (size_t)32 * n8 * sizeof(float),
                 hipMemcpyDeviceToDevice, stream);
  blockf(X8, K5, m8, 96, 64, 24, 24, 4, O8, T8, T8 + (size_t)64 * n8);

  convtr(O8, wt5, X4, 64, 32, 24, 24, 4);              bn(X4, m4, gt5, bt5, 32, n4, true);
  hipMemcpyAsync(X4 + (size_t)32 * n4, out_b2, (size_t)16 * n4 * sizeof(float),
                 hipMemcpyDeviceToDevice, stream);
  blockf(X4, K6, m4, 48, 32, 48, 48, 8, O4, T4, T4 + (size_t)32 * n4);

  convtr(O4, wt6, X2, 32, 16, 48, 48, 8);              bn(X2, m2, gt6, bt6, 16, n2, true);
  hipMemcpyAsync(X2 + (size_t)16 * n2, out_b1, (size_t)8 * n2 * sizeof(float),
                 hipMemcpyDeviceToDevice, stream);
  blockf(X2, K7, m2, 24, 16, 96, 96, 16, O2, T2, T2 + (size_t)16 * n2);

  convtr(O2, wt7, X1, 16, 8, 96, 96, 16);              bn(X1, m1, gt7, bt7, 8, n1, true);
  hipMemcpyAsync(X1 + (size_t)8 * n1, out_p1, (size_t)8 * n1 * sizeof(float),
                 hipMemcpyDeviceToDevice, stream);
  blockf(X1, K8, m1, 16, 8, GX, GY, GZ, O1, T1, T1 + (size_t)8 * n1);

  // ---- head + render ----
  conv(O1, wf, fin, 8, 2, GX, GY, GZ, 1, 1);
  k_bias_mask<<<dim3(gs(n1, 256), 2), 256, 0, stream>>>(fin, bf, m1, n1);
  k_render<<<gs(Nrays, 256), 256, 0, stream>>>(fin, origin, out_pts, tindex,
                                               (float*)d_out, Nrays);
  (void)out_size;
}
